// Attention_block_38233798869369
// MI455X (gfx1250) — matrix operand order repaired and checked by reading, not compiled, not run on hardware
//
#include <hip/hip_runtime.h>

#define SQ   2048
#define DMODEL 256
#define NH   4
#define DH   64
#define LDK  72   // padded LDS row stride (elements): 144B; matches TDM pad(32dw + 4dw)

typedef __attribute__((ext_vector_type(16))) __bf16 v16bf;
typedef __attribute__((ext_vector_type(8)))  float  v8f;
typedef __attribute__((ext_vector_type(4)))  unsigned int v4u;
typedef __attribute__((ext_vector_type(8)))  int    v8i;
typedef __attribute__((ext_vector_type(4)))  int    v4i;

union Frag16 {
    v4u q[2];
    unsigned short u[16];
    v16bf v;
};

__device__ __forceinline__ unsigned short f2bf(float f) {
    union { float f; unsigned u; } c; c.f = f;
    unsigned u = c.u;
    unsigned r = (u + 0x7FFFu + ((u >> 16) & 1u)) >> 16;
    return (unsigned short)r;
}

// A-fragment (16x32 bf16) from row-major [*, ld] source.
__device__ __forceinline__ Frag16 load_A(const unsigned short* src, int row0, int k0, int ld) {
    int l  = threadIdx.x & 31;
    int m  = row0 + (l & 15);
    int h8 = (l >> 4) << 3;
    const unsigned short* p = src + (size_t)m * ld + k0;
    Frag16 f;
    f.q[0] = *(const v4u*)(p + h8);
    f.q[1] = *(const v4u*)(p + 16 + h8);
    return f;
}

// B-fragment (32x16 bf16) from an N-major (transposed) source: srcT[n*ld + k].
__device__ __forceinline__ Frag16 load_B(const unsigned short* srcT, int n0, int k0, int ld) {
    int l = threadIdx.x & 31;
    const unsigned short* p = srcT + (size_t)(n0 + (l & 15)) * ld + k0 + ((l >> 4) << 3);
    Frag16 f;
    f.q[0] = *(const v4u*)(p);
    f.q[1] = *(const v4u*)(p + 16);
    return f;
}

__device__ __forceinline__ v8f wmma_bf16(const Frag16& a, const Frag16& b, v8f c) {
    return __builtin_amdgcn_wmma_f32_16x16x32_bf16(false, a.v, false, b.v, (short)0, c, false, false);
}

// ---------------------------------------------------------------------------
// TDM: one tensor_load_to_lds moves a 64x64 bf16 tile (row-major, element
// stride0 between rows) into LDS with a 16B pad after every 128B row
// (pad_interval=4 -> 32 DWORDs, pad_amount=3 -> 4 DWORDs) => LDS stride 72 elems.
// D# layout per CDNA5 ISA. Groups 2/3 zero (2-D).
// 6-arg builtin flavor (clang-23 / therock headers).
// ---------------------------------------------------------------------------
__device__ __forceinline__ void tdm_load_tile(const unsigned short* gtile,
                                              unsigned ldsByteOff, unsigned stride0) {
    unsigned long long ga = (unsigned long long)(size_t)gtile;
    v4u g0;
    g0[0] = 1u;                                               // count=1, user mode
    g0[1] = ldsByteOff;                                       // lds_addr
    g0[2] = (unsigned)(ga & 0xFFFFFFFFu);                     // global_addr[31:0]
    g0[3] = (unsigned)((ga >> 32) & 0x1FFFFFFu) | (2u << 30); // addr[56:32] | type=2
    v8i g1;
    g1[0] = (int)((1u << 16) | (1u << 20) | (4u << 22) | (3u << 25)); // 2B, pad 32dw+4dw
    g1[1] = (int)(64u << 16);            // tensor_dim0 = 64 (low 16 in [31:16])
    g1[2] = (int)(64u << 16);            // tensor_dim0 hi=0 | tensor_dim1=64 lo
    g1[3] = (int)(64u << 16);            // tensor_dim1 hi=0 | tile_dim0=64
    g1[4] = 64;                          // tile_dim1=64, tile_dim2=0
    g1[5] = (int)stride0;                // tensor_dim0_stride low 32 (elements)
    g1[6] = 0;                           // stride0 hi | tensor_dim1_stride lo
    g1[7] = 0;
    v4i z4 = {0, 0, 0, 0};
    v8i z8 = {0, 0, 0, 0, 0, 0, 0, 0};
    __builtin_amdgcn_tensor_load_to_lds(g0, g1, z4, z4, z8, 0);
}

// ---------------------------------------------------------------------------
// Kernel 0: fp32 -> bf16 pack. Weights stored TRANSPOSED (N-major).
// ---------------------------------------------------------------------------
__global__ void pack_kernel(const float* __restrict__ X,
                            const float* __restrict__ Wq, const float* __restrict__ Wk,
                            const float* __restrict__ Wv, const float* __restrict__ Wo,
                            unsigned short* __restrict__ Xbf,
                            unsigned short* __restrict__ WT4) {
    const size_t NX = (size_t)8 * SQ * DMODEL;
    const size_t NW = (size_t)DMODEL * DMODEL;
    const float* Ws[4] = {Wq, Wk, Wv, Wo};
    size_t total = NX + 4 * NW;
    size_t stride = (size_t)gridDim.x * blockDim.x;
    for (size_t i = (size_t)blockIdx.x * blockDim.x + threadIdx.x; i < total; i += stride) {
        if (i < NX) {
            Xbf[i] = f2bf(X[i]);
        } else {
            size_t j = i - NX;
            int w = (int)(j >> 16);
            int r = (int)(j & 65535);
            int k = r >> 8;
            int n = r & 255;
            WT4[(size_t)w * NW + (size_t)n * DMODEL + k] = f2bf(Ws[w][(size_t)k * DMODEL + n]);
        }
    }
}

// ---------------------------------------------------------------------------
// Kernel 1: QKV projection. grid = (BS/16, 1, 3); block = 128 (4 waves).
// ---------------------------------------------------------------------------
__global__ __launch_bounds__(128) void qkv_kernel(
        const unsigned short* __restrict__ Xbf, const unsigned short* __restrict__ WT4,
        const float* __restrict__ bq, const float* __restrict__ bk, const float* __restrict__ bv,
        unsigned short* __restrict__ Qbf, unsigned short* __restrict__ Kbf,
        unsigned short* __restrict__ Vtbf) {
    int mode = blockIdx.z;
    const unsigned short* WT = WT4 + (size_t)mode * DMODEL * DMODEL;
    const float* bias = (mode == 0) ? bq : ((mode == 1) ? bk : bv);
    int wave = threadIdx.x >> 5;
    int lane = threadIdx.x & 31;
    int m0 = blockIdx.x * 16;
    int n0w = wave * 64;

    v8f acc[4] = {};
#pragma unroll
    for (int kk = 0; kk < DMODEL; kk += 32) {
        Frag16 a = load_A(Xbf, m0, kk, DMODEL);
#pragma unroll
        for (int t = 0; t < 4; ++t) {
            Frag16 b = load_B(WT, n0w + t * 16, kk, DMODEL);
            acc[t] = wmma_bf16(a, b, acc[t]);
        }
    }

    int halfrow = (lane >> 4) << 3;
    int coll = lane & 15;
#pragma unroll
    for (int t = 0; t < 4; ++t) {
        int n = n0w + t * 16 + coll;
        float bn = bias[n];
        int h = n >> 6;
        int d = n & 63;
#pragma unroll
        for (int r = 0; r < 8; ++r) {
            int m = m0 + halfrow + r;
            int b = m >> 11;
            int s = m & (SQ - 1);
            unsigned short val = f2bf(acc[t][r] + bn);
            if (mode == 2) {
                Vtbf[(((size_t)(b * NH + h) * DH + d) * SQ) + s] = val;
            } else if (mode == 1) {
                Kbf[(((size_t)(b * NH + h) * SQ + s) * DH) + d] = val;
            } else {
                Qbf[(((size_t)(b * NH + h) * SQ + s) * DH) + d] = val;
            }
        }
    }
}

// ---------------------------------------------------------------------------
// Kernel 2: flash attention. grid = (S/64, B*H); block = 128 (4 waves).
// K/V tiles double-buffered in LDS via TDM (wave 0 issues, TENSORcnt-tracked).
// ---------------------------------------------------------------------------
__global__ __launch_bounds__(128) void attn_kernel(
        const unsigned short* __restrict__ Qbf, const unsigned short* __restrict__ Kbf,
        const unsigned short* __restrict__ Vtbf, const int* __restrict__ M,
        unsigned short* __restrict__ Obf) {
    __shared__ unsigned short ktile[2][64 * LDK];   // [kseq][d], padded
    __shared__ unsigned short vtile[2][64 * LDK];   // [d][kseq], padded
    __shared__ unsigned short pshare[4 * 16 * 64];  // per-wave 16x64 bf16 P tile

    int wave = threadIdx.x >> 5;
    int lane = threadIdx.x & 31;
    int bh = blockIdx.y;
    int b = bh >> 2;
    int h = bh & 3;
    size_t base = (size_t)bh * SQ * DH;
    const unsigned short* Qp  = Qbf + base;
    const unsigned short* Kp  = Kbf + base;
    const unsigned short* Vtp = Vtbf + base;
    unsigned short* lds = pshare + wave * (16 * 64);
    const int* Mb = M + (size_t)b * SQ * SQ;

    int q0 = blockIdx.x * 64 + wave * 16;
    int halfrow = (lane >> 4) << 3;
    int coll = lane & 15;

    Frag16 qf0 = load_A(Qp, q0, 0, DH);
    Frag16 qf1 = load_A(Qp, q0, 32, DH);

    v8f occ[4] = {};
    float mrow[8], lrow[8];
#pragma unroll
    for (int r = 0; r < 8; ++r) { mrow[r] = -3.0e38f; lrow[r] = 0.0f; }
    const float NEG = -3.0e38f;

    if (wave == 0) {                                    // prefetch tile 0
        tdm_load_tile(Kp,  (unsigned)(size_t)&ktile[0][0], DH);
        tdm_load_tile(Vtp, (unsigned)(size_t)&vtile[0][0], SQ);
    }

    for (int k0 = 0; k0 < SQ; k0 += 64) {
        int cur = (k0 >> 6) & 1;
        if (wave == 0) {
            if (k0 + 64 < SQ) {                         // prefetch tile k0+64
                tdm_load_tile(Kp + (size_t)(k0 + 64) * DH,
                              (unsigned)(size_t)&ktile[cur ^ 1][0], DH);
                tdm_load_tile(Vtp + (k0 + 64),
                              (unsigned)(size_t)&vtile[cur ^ 1][0], SQ);
                __builtin_amdgcn_s_wait_tensorcnt(2);   // tile k0 resident
            } else {
                __builtin_amdgcn_s_wait_tensorcnt(0);
            }
        }
        __syncthreads();   // publish tile k0 to all waves

        const unsigned short* ldsK = ktile[cur];
        const unsigned short* ldsV = vtile[cur];

        // scores: 16x64 tile = 4 C fragments, B-frags from LDS
        v8f sc[4] = {};
#pragma unroll
        for (int t = 0; t < 4; ++t) {
            Frag16 b0 = load_B(ldsK, t * 16, 0, LDK);
            sc[t] = wmma_bf16(qf0, b0, sc[t]);
            Frag16 b1 = load_B(ldsK, t * 16, 32, LDK);
            sc[t] = wmma_bf16(qf1, b1, sc[t]);
        }
        // scale + mask
#pragma unroll
        for (int t = 0; t < 4; ++t) {
            int kc = k0 + t * 16 + coll;
#pragma unroll
            for (int r = 0; r < 8; ++r) {
                int q = q0 + halfrow + r;
                float s = sc[t][r] * 0.125f;                 // 1/sqrt(64)
                if (Mb[(size_t)q * SQ + kc] == 0) s = NEG;
                sc[t][r] = s;
            }
        }
        // online softmax per row (16 lanes of a half-wave share a row)
#pragma unroll
        for (int r = 0; r < 8; ++r) {
            float mx = fmaxf(fmaxf(sc[0][r], sc[1][r]), fmaxf(sc[2][r], sc[3][r]));
#pragma unroll
            for (int msk = 1; msk <= 8; msk <<= 1) mx = fmaxf(mx, __shfl_xor(mx, msk, 32));
            float newm = fmaxf(mrow[r], mx);
            float safem = (newm < -1e37f) ? 0.0f : newm;
            float corr = __expf(mrow[r] - safem);
            float rs = 0.0f;
            int m = halfrow + r;
#pragma unroll
            for (int t = 0; t < 4; ++t) {
                float p = __expf(sc[t][r] - safem);
                rs += p;
                lds[m * 64 + t * 16 + coll] = f2bf(p);
            }
#pragma unroll
            for (int msk = 1; msk <= 8; msk <<= 1) rs += __shfl_xor(rs, msk, 32);
            lrow[r] = lrow[r] * corr + rs;
            mrow[r] = newm;
#pragma unroll
            for (int dt = 0; dt < 4; ++dt) occ[dt][r] = occ[dt][r] * corr;
        }
        // O += P * V   (P via per-wave LDS reshape; V B-frags from LDS tile)
        Frag16 pf0 = load_A(lds, 0, 0, 64);
        Frag16 pf1 = load_A(lds, 0, 32, 64);
#pragma unroll
        for (int dt = 0; dt < 4; ++dt) {
            Frag16 bv0 = load_B(ldsV, dt * 16, 0, LDK);
            occ[dt] = wmma_bf16(pf0, bv0, occ[dt]);
            Frag16 bv1 = load_B(ldsV, dt * 16, 32, LDK);
            occ[dt] = wmma_bf16(pf1, bv1, occ[dt]);
        }
        __syncthreads();   // done reading this buffer before TDM refills it
    }
    // normalize + merged-head store (bf16, row = b*S+s, col = h*64+d)
#pragma unroll
    for (int dt = 0; dt < 4; ++dt) {
#pragma unroll
        for (int r = 0; r < 8; ++r) {
            int s = q0 + halfrow + r;
            float v = occ[dt][r] / lrow[r];
            Obf[((size_t)(b * SQ + s)) * DMODEL + h * 64 + dt * 16 + coll] = f2bf(v);
        }
    }
}

// ---------------------------------------------------------------------------
// Kernel 3: output projection  out = Obf * Wo + bo   (fp32 store)
// ---------------------------------------------------------------------------
__global__ __launch_bounds__(128) void proj_kernel(
        const unsigned short* __restrict__ Obf, const unsigned short* __restrict__ WoT,
        const float* __restrict__ bo, float* __restrict__ out) {
    int wave = threadIdx.x >> 5;
    int lane = threadIdx.x & 31;
    int m0 = blockIdx.x * 16;
    int n0w = wave * 64;

    v8f acc[4] = {};
#pragma unroll
    for (int kk = 0; kk < DMODEL; kk += 32) {
        Frag16 a = load_A(Obf, m0, kk, DMODEL);
#pragma unroll
        for (int t = 0; t < 4; ++t) {
            Frag16 b = load_B(WoT, n0w + t * 16, kk, DMODEL);
            acc[t] = wmma_bf16(a, b, acc[t]);
        }
    }
    int halfrow = (lane >> 4) << 3;
    int coll = lane & 15;
#pragma unroll
    for (int t = 0; t < 4; ++t) {
        int n = n0w + t * 16 + coll;
        float bn = bo[n];
#pragma unroll
        for (int r = 0; r < 8; ++r) {
            int m = m0 + halfrow + r;
            out[(size_t)m * DMODEL + n] = acc[t][r] + bn;
        }
    }
}

extern "C" void kernel_launch(void* const* d_in, const int* in_sizes, int n_in,
                              void* d_out, int out_size, void* d_ws, size_t ws_size,
                              hipStream_t stream) {
    const float* X  = (const float*)d_in[0];
    const int*   M  = (const int*)d_in[1];
    const float* Wq = (const float*)d_in[2];
    const float* bq = (const float*)d_in[3];
    const float* Wk = (const float*)d_in[4];
    const float* bk = (const float*)d_in[5];
    const float* Wv = (const float*)d_in[6];
    const float* bv = (const float*)d_in[7];
    const float* Wo = (const float*)d_in[8];
    const float* bo = (const float*)d_in[9];

    const size_t NX = (size_t)8 * SQ * DMODEL;
    const size_t NW = (size_t)DMODEL * DMODEL;

    unsigned short* Xbf  = (unsigned short*)d_ws;
    unsigned short* WT4  = Xbf + NX;
    unsigned short* Qbf  = WT4 + 4 * NW;
    unsigned short* Kbf  = Qbf + NX;
    unsigned short* Vtbf = Kbf + NX;
    unsigned short* Obf  = Vtbf + NX;

    pack_kernel<<<2048, 256, 0, stream>>>(X, Wq, Wk, Wv, Wo, Xbf, WT4);

    dim3 g1((8 * SQ) / 16, 1, 3);
    qkv_kernel<<<g1, 128, 0, stream>>>(Xbf, WT4, bq, bk, bv, Qbf, Kbf, Vtbf);

    dim3 g2(SQ / 64, 8 * NH);
    attn_kernel<<<g2, 128, 0, stream>>>(Qbf, Kbf, Vtbf, M, Obf);

    proj_kernel<<<(8 * SQ) / 16, 128, 0, stream>>>(Obf, WT4 + 3 * NW, bo, (float*)d_out);
}
